// SpatialBasedGraphConvNet_37280316129400
// MI455X (gfx1250) — compile-verified, matrix operand order checked by reading, not run on hardware
//
#include <hip/hip_runtime.h>
#include <hip/hip_bf16.h>
#include <math.h>

// ---------------------------------------------------------------------------
// SpatialBasedGraphConvNet for MI455X (gfx1250, wave32, WMMA)
//
//  support = x @ gc_w                     (WMMA, epilogue -> f16 transposed)
//  h       = adj @ support + gc_b        (WMMA, split-K, atomic f32 combine)
//  z       = tanh(h @ mlp_w + mlp_b)     (WMMA, epilogue -> zcat slice)
//  out     = zcat @ cls_w + cls_b        (tiny, VALU)
// ---------------------------------------------------------------------------

typedef __attribute__((ext_vector_type(16))) _Float16 v16h;
typedef __attribute__((ext_vector_type(8)))  _Float16 v8h;
typedef __attribute__((ext_vector_type(4)))  _Float16 v4h;
typedef __attribute__((ext_vector_type(8)))  float    v8f;
typedef __attribute__((ext_vector_type(4)))  float    v4f;

#define NROWS  6144
#define CIN    227
#define NCLS   27

// Build a 16x16x32-f16 WMMA fragment from two contiguous 8-half chunks
// (K = hi*8..hi*8+7 in regs 0..3, K = 16+hi*8.. in regs 4..7 per ISA layout).
__device__ __forceinline__ v16h mkfrag(const _Float16* p) {
    v8h a = *(const v8h*)(p);
    v8h b = *(const v8h*)(p + 16);
    v16h r;
#pragma unroll
    for (int i = 0; i < 8; ++i) { r[i] = a[i]; r[i + 8] = b[i]; }
    return r;
}

// Generic tiled GEMM:  C[BM x BN] (+ epilogue) = A_f32[M x K] * BT_f16[BN x K]^T
//   MODE 0: store (f16)C transposed into outT[BN_total=BN][NROWS]  (support)
//   MODE 1: atomicAdd f32 into outAcc[M][BN]                       (split-K agg)
//   MODE 2: outZ[row][zOff+col] = tanh(C + bias[col]) for col < NE (MLP)
template <int BN, int WROWS, int WCOLS, int MODE, int NE>
__global__ __launch_bounds__(256)
void gemm_wmma(const float* __restrict__ A,
               const _Float16* __restrict__ BT,
               int K, int kChunk,
               const float* __restrict__ bias,
               _Float16* __restrict__ outT,
               float* __restrict__ outAcc,
               float* __restrict__ outZ, int zLd, int zOff)
{
    constexpr int BM  = 128;
    constexpr int BK  = 32;
    constexpr int WMT = BM / WROWS / 16;   // 16x16 row tiles per wave
    constexpr int WNT = BN / WCOLS / 16;   // 16x16 col tiles per wave
    constexpr int AP  = BK + 8;            // 40 halves -> conflict-free b128 reads
    constexpr int BP  = BK + 8;

    __shared__ _Float16 Alds[BM * AP];
    __shared__ _Float16 Blds[BN * BP];

    const int tid  = threadIdx.x;
    const int lane = tid & 31;
    const int wave = tid >> 5;
    const int lo   = lane & 15;
    const int hi   = lane >> 4;
    const int wr   = wave / WCOLS;
    const int wc   = wave % WCOLS;
    const int waveRow = wr * WMT * 16;
    const int waveCol = wc * WNT * 16;
    const int rowBase = blockIdx.x * BM;
    const int kBeg = blockIdx.y * kChunk;
    const int kEnd = kBeg + kChunk;

    v8f acc[WMT][WNT] = {};

    for (int k0 = kBeg; k0 < kEnd; k0 += BK) {
        // ---- A tile: 128x32 f32 -> NaN-fill -> f16 -> LDS ----
#pragma unroll
        for (int it = 0; it < 4; ++it) {
            int linear = tid + it * 256;           // 1024 v4f chunks
            int r = linear >> 3;                   // 8 chunks per row
            int c = (linear & 7) << 2;
            v4f av = *(const v4f*)(A + (size_t)(rowBase + r) * K + k0 + c);
            v4h hv;
#pragma unroll
            for (int q = 0; q < 4; ++q) {
                float x = av[q];
                x = (x == x) ? x : 0.0f;           // jnp.where(isnan, 0, .)
                hv[q] = (_Float16)x;
            }
            *(v4h*)(Alds + r * AP + c) = hv;
        }
        // ---- B tile: BN x 32 f16 (BT is [BN_total][K], row-contiguous) ----
        for (int linear = tid; linear < BN * 4; linear += 256) {
            int n = linear >> 2;
            int c = (linear & 3) << 3;
            v8h bv = *(const v8h*)(BT + (size_t)n * K + k0 + c);
            *(v8h*)(Blds + n * BP + c) = bv;
        }
        // prefetch next A tile while this one is consumed
        if (k0 + BK < kEnd)
            __builtin_prefetch(A + (size_t)(rowBase + (tid >> 1)) * K + k0 + BK, 0, 1);
        __syncthreads();

        v16h aF[WMT], bF[WNT];
#pragma unroll
        for (int i = 0; i < WMT; ++i)
            aF[i] = mkfrag(Alds + (waveRow + i * 16 + lo) * AP + hi * 8);
#pragma unroll
        for (int j = 0; j < WNT; ++j)
            bF[j] = mkfrag(Blds + (waveCol + j * 16 + lo) * BP + hi * 8);
#pragma unroll
        for (int i = 0; i < WMT; ++i)
#pragma unroll
            for (int j = 0; j < WNT; ++j)
                acc[i][j] = __builtin_amdgcn_wmma_f32_16x16x32_f16(
                    false, aF[i], false, bF[j], (short)0, acc[i][j], false, false);
        __syncthreads();
    }

    // ---- epilogues (C/D layout: col = lo, row = r + 8*hi within 16x16 tile) ----
#pragma unroll
    for (int i = 0; i < WMT; ++i) {
#pragma unroll
        for (int j = 0; j < WNT; ++j) {
            const int col = waveCol + j * 16 + lo;
#pragma unroll
            for (int r = 0; r < 8; ++r) {
                const int row = rowBase + waveRow + i * 16 + r + hi * 8;
                float v = acc[i][j][r];
                if constexpr (MODE == 0) {
                    outT[(size_t)col * NROWS + row] = (_Float16)v;
                } else if constexpr (MODE == 1) {
                    atomicAdd(outAcc + (size_t)row * BN + col, v);
                } else {
                    if (col < NE)
                        outZ[(size_t)row * zLd + zOff + col] = tanhf(v + bias[col]);
                }
            }
        }
    }
}

// Convert W[K][Nout] f32 -> WT[NoutPad][K] f16, zero-padding rows >= Nout.
__global__ void prep_wT(const float* __restrict__ src, _Float16* __restrict__ dst,
                        int K, int Nout, int NoutPad)
{
    int idx = blockIdx.x * blockDim.x + threadIdx.x;
    if (idx >= K * NoutPad) return;
    int o = idx / K;
    int k = idx - o * K;
    float v = (o < Nout) ? src[(size_t)k * Nout + o] : 0.0f;
    dst[(size_t)o * K + k] = (_Float16)v;
}

// h[r][c] = gc_b[c]  (split-K GEMM atomically accumulates on top)
__global__ void init_h(float* __restrict__ h, const float* __restrict__ b,
                       int nh, int total)
{
    int i = blockIdx.x * blockDim.x + threadIdx.x;
    if (i < total) h[i] = b[i % nh];
}

// out[r][c] = zcat[r][:] . cls_w[:][c] + cls_b[c]   (6144 x 27, tiny)
__global__ void cls_kernel(const float* __restrict__ zcat,
                           const float* __restrict__ w,
                           const float* __restrict__ b,
                           float* __restrict__ out)
{
    int idx = blockIdx.x * blockDim.x + threadIdx.x;
    if (idx >= NROWS * NCLS) return;
    int r = idx / NCLS;
    int c = idx - r * NCLS;
    const float* zr = zcat + (size_t)r * CIN;
    float acc = b[c];
#pragma unroll 4
    for (int k = 0; k < CIN; ++k) acc += zr[k] * w[k * NCLS + c];
    out[idx] = acc;
}

extern "C" void kernel_launch(void* const* d_in, const int* in_sizes, int n_in,
                              void* d_out, int out_size, void* d_ws, size_t ws_size,
                              hipStream_t stream)
{
    const float* x_s   = (const float*)d_in[0];
    const float* x_t   = (const float*)d_in[1];
    const float* x_g   = (const float*)d_in[2];
    const float* adj_s = (const float*)d_in[3];
    const float* adj_t = (const float*)d_in[4];
    const float* adj_g = (const float*)d_in[5];
    const float* gcw_s = (const float*)d_in[6];
    const float* gcb_s = (const float*)d_in[7];
    const float* gcw_t = (const float*)d_in[8];
    const float* gcb_t = (const float*)d_in[9];
    const float* gcw_g = (const float*)d_in[10];
    const float* gcb_g = (const float*)d_in[11];
    const float* mw_s  = (const float*)d_in[12];
    const float* mb_s  = (const float*)d_in[13];
    const float* mw_t  = (const float*)d_in[14];
    const float* mb_t  = (const float*)d_in[15];
    const float* mw_g  = (const float*)d_in[16];
    const float* mb_g  = (const float*)d_in[17];
    const float* clsw  = (const float*)d_in[18];
    const float* clsb  = (const float*)d_in[19];
    float* out = (float*)d_out;

    char* ws = (char*)d_ws;
    size_t off = 0;
    auto alloc = [&](size_t bytes) -> void* {
        void* p = ws + off;
        off = (off + bytes + 255) & ~(size_t)255;
        return p;
    };
    _Float16* gcwT_s = (_Float16*)alloc((size_t)256 * 224 * 2);
    _Float16* gcwT_t = (_Float16*)alloc((size_t)128 * 64 * 2);
    _Float16* gcwT_g = (_Float16*)alloc((size_t)64 * 32 * 2);
    _Float16* mwT_s  = (_Float16*)alloc((size_t)144 * 256 * 2);  // 129 -> 144 padded
    _Float16* mwT_t  = (_Float16*)alloc((size_t)80 * 128 * 2);   // 65 -> 80
    _Float16* mwT_g  = (_Float16*)alloc((size_t)48 * 64 * 2);    // 33 -> 48
    _Float16* supT_s = (_Float16*)alloc((size_t)256 * NROWS * 2);
    _Float16* supT_t = (_Float16*)alloc((size_t)128 * NROWS * 2);
    _Float16* supT_g = (_Float16*)alloc((size_t)64 * NROWS * 2);
    float*    h_s    = (float*)alloc((size_t)NROWS * 256 * 4);
    float*    h_t    = (float*)alloc((size_t)NROWS * 128 * 4);
    float*    h_g    = (float*)alloc((size_t)NROWS * 64 * 4);
    float*    zcat   = (float*)alloc((size_t)NROWS * CIN * 4);

    const int MT = NROWS / 128;   // 48 M-tiles
    const int KSPLIT = 8;         // split-K for the big aggregation GEMMs

    // --- stage 0: weight prep (f32 -> transposed f16, padded) + h init ---
    auto g1 = [](int total) { return dim3((total + 255) / 256); };
    prep_wT<<<g1(224 * 256), 256, 0, stream>>>(gcw_s, gcwT_s, 224, 256, 256);
    prep_wT<<<g1(64 * 128), 256, 0, stream>>>(gcw_t, gcwT_t, 64, 128, 128);
    prep_wT<<<g1(32 * 64), 256, 0, stream>>>(gcw_g, gcwT_g, 32, 64, 64);
    prep_wT<<<g1(256 * 144), 256, 0, stream>>>(mw_s, mwT_s, 256, 129, 144);
    prep_wT<<<g1(128 * 80), 256, 0, stream>>>(mw_t, mwT_t, 128, 65, 80);
    prep_wT<<<g1(64 * 48), 256, 0, stream>>>(mw_g, mwT_g, 64, 33, 48);
    init_h<<<g1(NROWS * 256), 256, 0, stream>>>(h_s, gcb_s, 256, NROWS * 256);
    init_h<<<g1(NROWS * 128), 256, 0, stream>>>(h_t, gcb_t, 128, NROWS * 128);
    init_h<<<g1(NROWS * 64), 256, 0, stream>>>(h_g, gcb_g, 64, NROWS * 64);

    // --- stage 1: supportT = (x @ gc_w)^T as f16 ---
    gemm_wmma<256, 2, 4, 0, 0><<<dim3(MT, 1), 256, 0, stream>>>(
        x_s, gcwT_s, 224, 224, nullptr, supT_s, nullptr, nullptr, 0, 0);
    gemm_wmma<128, 2, 4, 0, 0><<<dim3(MT, 1), 256, 0, stream>>>(
        x_t, gcwT_t, 64, 64, nullptr, supT_t, nullptr, nullptr, 0, 0);
    gemm_wmma<64, 2, 4, 0, 0><<<dim3(MT, 1), 256, 0, stream>>>(
        x_g, gcwT_g, 32, 32, nullptr, supT_g, nullptr, nullptr, 0, 0);

    // --- stage 2: h += adj @ support (split-K, f32 atomic combine) ---
    gemm_wmma<256, 2, 4, 1, 0><<<dim3(MT, KSPLIT), 256, 0, stream>>>(
        adj_s, supT_s, NROWS, NROWS / KSPLIT, nullptr, nullptr, h_s, nullptr, 0, 0);
    gemm_wmma<128, 2, 4, 1, 0><<<dim3(MT, KSPLIT), 256, 0, stream>>>(
        adj_t, supT_t, NROWS, NROWS / KSPLIT, nullptr, nullptr, h_t, nullptr, 0, 0);
    gemm_wmma<64, 2, 4, 1, 0><<<dim3(MT, KSPLIT), 256, 0, stream>>>(
        adj_g, supT_g, NROWS, NROWS / KSPLIT, nullptr, nullptr, h_g, nullptr, 0, 0);

    // --- stage 3: zcat slices = tanh(h @ mlp_w + mlp_b) ---
    gemm_wmma<144, 8, 1, 2, 129><<<dim3(MT, 1), 256, 0, stream>>>(
        h_s, mwT_s, 256, 256, mb_s, nullptr, nullptr, zcat, CIN, 0);
    gemm_wmma<80, 8, 1, 2, 65><<<dim3(MT, 1), 256, 0, stream>>>(
        h_t, mwT_t, 128, 128, mb_t, nullptr, nullptr, zcat, CIN, 129);
    gemm_wmma<48, 8, 1, 2, 33><<<dim3(MT, 1), 256, 0, stream>>>(
        h_g, mwT_g, 64, 64, mb_g, nullptr, nullptr, zcat, CIN, 194);

    // --- stage 4: classifier ---
    cls_kernel<<<g1(NROWS * NCLS), 256, 0, stream>>>(zcat, clsw, clsb, out);
}
